// MaskedAttentionHead_1786706395435
// MI455X (gfx1250) — matrix operand order repaired and checked by reading, not compiled, not run on hardware
//
#include <hip/hip_runtime.h>
#include <math.h>

// ---------------- types -----------------
typedef __bf16 bf16_t;
typedef bf16_t v16bf __attribute__((ext_vector_type(16)));
typedef bf16_t v8bf  __attribute__((ext_vector_type(8)));
typedef float  v8f   __attribute__((ext_vector_type(8)));
typedef float  v4f   __attribute__((ext_vector_type(4)));
typedef unsigned int u32x4 __attribute__((ext_vector_type(4)));
typedef int          i32x8 __attribute__((ext_vector_type(8)));
typedef int          i32x4 __attribute__((ext_vector_type(4)));

#define BATCH 4
#define SEQ   4096
#define DIN   512
#define DOUT  64

#if __has_builtin(__builtin_amdgcn_tensor_load_to_lds) && \
    __has_builtin(__builtin_amdgcn_s_wait_tensorcnt)
#define HAVE_TDM 1
#else
#define HAVE_TDM 0
#endif

// ---- async global->LDS helpers (gfx1250, ASYNCcnt-tracked) ----
__device__ __forceinline__ void async_g2l_b128(unsigned lds_addr, unsigned goff,
                                               unsigned long long gbase) {
  asm volatile("global_load_async_to_lds_b128 %0, %1, %2"
               :: "v"(lds_addr), "v"(goff), "s"(gbase) : "memory");
}
__device__ __forceinline__ void wait_async_le2() {
  asm volatile("s_wait_asynccnt 2" ::: "memory");
}
__device__ __forceinline__ void wait_async_0() {
  asm volatile("s_wait_asynccnt 0" ::: "memory");
}
__device__ __forceinline__ unsigned lds_off(const void* p) {
  return (unsigned)(size_t)p;  // generic->LDS: addr[31:0] is the LDS byte offset
}

#if HAVE_TDM
// ---- Tensor Data Mover: one descriptor moves a whole 2D bf16 tile ----
// D# packing per CDNA5 ISA ch.8: group0 {count,lds_addr,global_addr,type=2},
// group1 {data_size=2B, tensor dims, tile dims, dim0 stride}; groups2/3 = 0
// (tile_dim2 = 0 -> 2D). Tracked with TENSORcnt.
// clang-23 builtin arity: (u32x4, i32x8, i32x4, i32x4, i32x8, i32 cpol)
__device__ __forceinline__ void tdm_load_2d(
    unsigned lds_byte, unsigned long long gaddr,
    unsigned tile0, unsigned tile1,              // tile extents (elements)
    unsigned long long stride0,                  // elems between dim1 lines
    unsigned tdim0, unsigned tdim1) {            // tensor extents (elements)
  u32x4 g0;
  g0[0] = 1u;                                        // count=1 (valid, user)
  g0[1] = lds_byte;                                  // lds_addr [63:32]
  g0[2] = (unsigned)(gaddr & 0xFFFFFFFFu);           // global_addr [95:64]
  g0[3] = (unsigned)((gaddr >> 32) & 0x01FFFFFFu)    // global_addr [120:96]
        | (2u << 30);                                // type=2 ("image")
  i32x8 g1;
  g1[0] = (int)(1u << 16);                           // data_size=1 -> 2 bytes
  g1[1] = (int)((tdim0 & 0xFFFFu) << 16);            // tensor_dim0 [79:48]
  g1[2] = (int)(((tdim0 >> 16) & 0xFFFFu) | ((tdim1 & 0xFFFFu) << 16));
  g1[3] = (int)(((tdim1 >> 16) & 0xFFFFu) | ((tile0 & 0xFFFFu) << 16));
  g1[4] = (int)(tile1 & 0xFFFFu);                    // tile_dim1; tile_dim2=0
  g1[5] = (int)(unsigned)(stride0 & 0xFFFFFFFFu);    // dim0 stride [207:160]
  g1[6] = (int)(unsigned)((stride0 >> 32) & 0xFFFFu);
  g1[7] = 0;
  i32x4 z4 = {0, 0, 0, 0};
  i32x8 z8 = {0, 0, 0, 0, 0, 0, 0, 0};
  __builtin_amdgcn_tensor_load_to_lds(g0, g1, z4, z4, z8, 0);
}
#endif

// --------------------------------------------------------------------------
// Kernel 0: fp32 weights [64,512] x3 -> bf16 workspace
// --------------------------------------------------------------------------
__global__ void cvt_weights(const float* __restrict__ wq,
                            const float* __restrict__ wk,
                            const float* __restrict__ wv,
                            bf16_t* __restrict__ out) {
  const int N = DOUT * DIN;
  int i = blockIdx.x * 256 + threadIdx.x;
  if (i < N)            out[i] = (bf16_t)wq[i];
  else if (i < 2 * N)   out[i] = (bf16_t)wk[i - N];
  else if (i < 3 * N)   out[i] = (bf16_t)wv[i - 2 * N];
}

// --------------------------------------------------------------------------
// Kernel 1: Y = X @ W^T via v_wmma_f32_16x16x32_bf16. Weight async-staged
// into LDS once per block (ASYNCcnt path); B operands via ds_load_b128.
// --------------------------------------------------------------------------
__global__ __launch_bounds__(128) void qkv_proj(
    const float* __restrict__ X, const bf16_t* __restrict__ Wb,
    bf16_t* __restrict__ Yrm, bf16_t* __restrict__ Yt, int transposed) {
  __shared__ __align__(64) bf16_t lds_w[DOUT * DIN];  // 64 KB

  const int lane = threadIdx.x & 31;
  const int wave = threadIdx.x >> 5;
  const int m    = lane & 15;
  const int hi   = lane >> 4;
  const int rowbase = blockIdx.x * 64 + wave * 16;

  {
    const unsigned long long gbase = (unsigned long long)(size_t)Wb;
    const unsigned lbase = lds_off(&lds_w[0]);
    unsigned o = threadIdx.x * 16;
#pragma unroll
    for (int it = 0; it < 32; ++it, o += 2048)
      async_g2l_b128(lbase + o, o, gbase);
    wait_async_0();
  }
  __syncthreads();

  const float* xrow = X + (size_t)(rowbase + m) * DIN;
  v8f acc[4] = {};

  for (int k0 = 0; k0 < DIN; k0 += 32) {
    const float* pa0 = xrow + k0 + hi * 16;
    const float* pa1 = xrow + k0 + hi * 16 + 8;
    v4f f0 = *(const v4f*)(pa0);
    v4f f1 = *(const v4f*)(pa0 + 4);
    v4f f2 = *(const v4f*)(pa1);
    v4f f3 = *(const v4f*)(pa1 + 4);
    v16bf a;
#pragma unroll
    for (int e = 0; e < 4; ++e) {
      a[e]      = (bf16_t)f0[e];
      a[e + 4]  = (bf16_t)f1[e];
      a[e + 8]  = (bf16_t)f2[e];
      a[e + 12] = (bf16_t)f3[e];
    }
#pragma unroll
    for (int t = 0; t < 4; ++t) {
      v16bf bm = *(const v16bf*)(lds_w + (size_t)(t * 16 + m) * DIN + k0 + hi * 16);
      acc[t] = __builtin_amdgcn_wmma_f32_16x16x32_bf16(
          false, a, false, bm, (short)0, acc[t], false, false);
    }
  }

  if (!transposed) {
#pragma unroll
    for (int t = 0; t < 4; ++t)
#pragma unroll
      for (int r = 0; r < 8; ++r)
        Yrm[(size_t)(rowbase + r + hi * 8) * DOUT + t * 16 + m] = (bf16_t)acc[t][r];
  } else {
    const int b  = rowbase / SEQ;
    const int s0 = (rowbase % SEQ) + hi * 8;
#pragma unroll
    for (int t = 0; t < 4; ++t) {
      const int d = t * 16 + m;
      v8bf pack;
#pragma unroll
      for (int r = 0; r < 8; ++r) pack[r] = (bf16_t)acc[t][r];
      *(v8bf*)(Yt + ((size_t)b * DOUT + d) * SEQ + s0) = pack;
    }
  }
}

// --------------------------------------------------------------------------
// Kernel 2: flash attention. K/V 32-key tiles staged to LDS by the TDM
// (one descriptor per tile, wave 0 issues, TENSORcnt double buffering);
// falls back to ASYNCcnt loads if the TDM builtin is unavailable.
// --------------------------------------------------------------------------
__global__ __launch_bounds__(256) void attn(
    const bf16_t* __restrict__ Qb, const bf16_t* __restrict__ Kb,
    const bf16_t* __restrict__ Vt, float* __restrict__ Out) {
  __shared__ __align__(64) bf16_t lds_k[2][32 * DOUT];  // 2 x 4KB, [key][d]
  __shared__ __align__(64) bf16_t lds_v[2][DOUT * 32];  // 2 x 4KB, [d][key]
  __shared__ __align__(64) bf16_t lds_p[8][16 * 32];    // per-wave P tile

  const int tid  = threadIdx.x;
  const int lane = tid & 31;
  const int wave = tid >> 5;
  const int m    = lane & 15;
  const int hi   = lane >> 4;
  const int b    = blockIdx.y;
  const int qblk = blockIdx.x * 128;
  const int qbase = qblk + wave * 16;

  // Q in A-layout (loaded once)
  const bf16_t* qrow = Qb + ((size_t)b * SEQ + qbase + m) * DOUT;
  v16bf qa[2];
#pragma unroll
  for (int c = 0; c < 2; ++c) {
    v8bf lo = *(const v8bf*)(qrow + c * 32 + hi * 16);
    v8bf up = *(const v8bf*)(qrow + c * 32 + hi * 16 + 8);
#pragma unroll
    for (int e = 0; e < 8; ++e) { qa[c][e] = lo[e]; qa[c][e + 8] = up[e]; }
  }

  v8f Oacc[4] = {};
  float rmax[8], rsum[8];
#pragma unroll
  for (int r = 0; r < 8; ++r) { rmax[r] = -__builtin_inff(); rsum[r] = 0.f; }
  const float inv_scale = 0.125f;               // 1/sqrt(64)
  const float NEGBIG = -3.4028234663852886e38f; // finfo(f32).min

  const unsigned long long kbb = (unsigned long long)(size_t)(Kb + (size_t)b * SEQ * DOUT);
  const unsigned long long vbb = (unsigned long long)(size_t)(Vt + (size_t)b * DOUT * SEQ);
#if !HAVE_TDM
  const int vrow = tid >> 2, vc = tid & 3;
#endif

  auto stage = [&](int j0, int buf) {
#if HAVE_TDM
    if (wave == 0) {
      // K tile: 32 seq-rows x 64 d (contiguous rows, stride 64 elems)
      tdm_load_2d(lds_off(&lds_k[buf][0]),
                  kbb + (unsigned long long)j0 * DOUT * 2,
                  /*tile0=*/DOUT, /*tile1=*/32, /*stride0=*/DOUT,
                  /*tdim0=*/DOUT, /*tdim1=*/SEQ);
      // V tile: 64 d-rows x 32 keys (row stride SEQ elems)
      tdm_load_2d(lds_off(&lds_v[buf][0]),
                  vbb + (unsigned long long)j0 * 2,
                  /*tile0=*/32, /*tile1=*/DOUT, /*stride0=*/SEQ,
                  /*tdim0=*/SEQ, /*tdim1=*/DOUT);
    }
#else
    async_g2l_b128(lds_off(&lds_k[buf][0]) + tid * 16,
                   (unsigned)(j0 * DOUT * 2) + tid * 16, kbb);
    async_g2l_b128(lds_off(&lds_v[buf][0]) + vrow * 64 + vc * 16,
                   (unsigned)(vrow * SEQ * 2 + j0 * 2 + vc * 16), vbb);
#endif
  };

  const int ntiles = (qblk + 128) / 32;  // uniform across the block
  stage(0, 0);

  for (int it = 0; it < ntiles; ++it) {
    const int j0 = it * 32;
    const int cur = it & 1;
    if (it + 1 < ntiles) {
      stage(j0 + 32, cur ^ 1);
#if HAVE_TDM
      if (wave == 0) __builtin_amdgcn_s_wait_tensorcnt(2);  // cur tile landed
#else
      wait_async_le2();
#endif
    } else {
#if HAVE_TDM
      if (wave == 0) __builtin_amdgcn_s_wait_tensorcnt(0);
#else
      wait_async_0();
#endif
    }
    __syncthreads();  // current tile resident for every wave

    // ---- S tile = Q K^T from LDS ----
    v8f s[2] = {};
#pragma unroll
    for (int sub = 0; sub < 2; ++sub) {
      const bf16_t* krow = &lds_k[cur][(sub * 16 + m) * DOUT + hi * 16];
      v16bf kb0 = *(const v16bf*)(krow);
      v16bf kb1 = *(const v16bf*)(krow + 32);
      s[sub] = __builtin_amdgcn_wmma_f32_16x16x32_bf16(
          false, qa[0], false, kb0, (short)0, s[sub], false, false);
      s[sub] = __builtin_amdgcn_wmma_f32_16x16x32_bf16(
          false, qa[1], false, kb1, (short)0, s[sub], false, false);
    }

    // ---- mask + online softmax ----
    float x0[8], x1[8], nmax[8];
#pragma unroll
    for (int r = 0; r < 8; ++r) {
      const int irow = qbase + r + 8 * hi;
      float a0 = ((j0 + m) >= irow)      ? NEGBIG : s[0][r];
      float a1 = ((j0 + 16 + m) >= irow) ? NEGBIG : s[1][r];
      a0 *= inv_scale; a1 *= inv_scale;
      float mx = fmaxf(a0, a1);
#pragma unroll
      for (int off = 1; off < 16; off <<= 1) mx = fmaxf(mx, __shfl_xor(mx, off, 32));
      nmax[r] = fmaxf(mx, rmax[r]);
      x0[r] = a0; x1[r] = a1;
    }
#pragma unroll
    for (int r = 0; r < 8; ++r) {
      const float alpha = __expf(rmax[r] - nmax[r]);
      rmax[r] = nmax[r];
      const float p0 = __expf(x0[r] - nmax[r]);
      const float p1 = __expf(x1[r] - nmax[r]);
      float ps = p0 + p1;
#pragma unroll
      for (int off = 1; off < 16; off <<= 1) ps += __shfl_xor(ps, off, 32);
      rsum[r] = rsum[r] * alpha + ps;
#pragma unroll
      for (int t = 0; t < 4; ++t) Oacc[t][r] *= alpha;
      lds_p[wave][(r + 8 * hi) * 32 + m]      = (bf16_t)p0;
      lds_p[wave][(r + 8 * hi) * 32 + 16 + m] = (bf16_t)p1;
    }

    // ---- P: D-layout -> A-layout via per-wave LDS round trip ----
    v16bf pa;
    {
      const bf16_t* base = &lds_p[wave][0];
      v8bf lo = *(const v8bf*)(base + m * 32 + hi * 16);
      v8bf up = *(const v8bf*)(base + m * 32 + hi * 16 + 8);
#pragma unroll
      for (int e = 0; e < 8; ++e) { pa[e] = lo[e]; pa[e + 8] = up[e]; }
    }

    // ---- O += P V from LDS ----
#pragma unroll
    for (int t = 0; t < 4; ++t) {
      v16bf vb = *(const v16bf*)(&lds_v[cur][(t * 16 + m) * 32 + hi * 16]);
      Oacc[t] = __builtin_amdgcn_wmma_f32_16x16x32_bf16(
          false, pa, false, vb, (short)0, Oacc[t], false, false);
    }

    __syncthreads();  // everyone done reading buf before it is re-staged
  }

  // ---- epilogue ----
#pragma unroll
  for (int r = 0; r < 8; ++r) {
    const float invl = 1.0f / rsum[r];
    float* orow = Out + ((size_t)b * SEQ + qbase + r + 8 * hi) * DOUT;
#pragma unroll
    for (int t = 0; t < 4; ++t) orow[t * 16 + m] = Oacc[t][r] * invl;
  }
}

// --------------------------------------------------------------------------
// Kernel 3: row i=0 is fully masked -> uniform softmax over all S keys
// --------------------------------------------------------------------------
__global__ void row0_fix(const bf16_t* __restrict__ Vt, float* __restrict__ Out) {
  const int b = blockIdx.x;
  const int d = threadIdx.x;  // 64 threads
  float s = 0.f;
  const bf16_t* col = Vt + ((size_t)b * DOUT + d) * SEQ;
  for (int j = 0; j < SEQ; ++j) s += (float)col[j];
  Out[(size_t)b * SEQ * DOUT + d] = s * (1.0f / (float)SEQ);
}

// --------------------------------------------------------------------------
extern "C" void kernel_launch(void* const* d_in, const int* in_sizes, int n_in,
                              void* d_out, int out_size, void* d_ws, size_t ws_size,
                              hipStream_t stream) {
  const float* query = (const float*)d_in[0];
  const float* key_t = (const float*)d_in[1];
  const float* value = (const float*)d_in[2];
  const float* Wq    = (const float*)d_in[3];
  const float* Wk    = (const float*)d_in[4];
  const float* Wv    = (const float*)d_in[5];
  float* out = (float*)d_out;

  char* ws = (char*)d_ws;
  const size_t WBYTES = (size_t)DOUT * DIN * sizeof(bf16_t);          // 64 KB
  const size_t QBYTES = (size_t)BATCH * SEQ * DOUT * sizeof(bf16_t);  // 2 MB
  bf16_t* WqB = (bf16_t*)(ws);
  bf16_t* WkB = (bf16_t*)(ws + WBYTES);
  bf16_t* WvB = (bf16_t*)(ws + 2 * WBYTES);
  bf16_t* Qb  = (bf16_t*)(ws + 3 * WBYTES);
  bf16_t* Kb  = (bf16_t*)(ws + 3 * WBYTES + QBYTES);
  bf16_t* Vt  = (bf16_t*)(ws + 3 * WBYTES + 2 * QBYTES);

  cvt_weights<<<(3 * DOUT * DIN + 255) / 256, 256, 0, stream>>>(Wq, Wk, Wv, WqB);

  const int nblk = BATCH * SEQ / 64;  // 256
  qkv_proj<<<nblk, 128, 0, stream>>>(query, WqB, Qb, nullptr, 0);
  qkv_proj<<<nblk, 128, 0, stream>>>(key_t, WkB, Kb, nullptr, 0);
  qkv_proj<<<nblk, 128, 0, stream>>>(value, WvB, nullptr, Vt, 1);

  dim3 agrid(SEQ / 128, BATCH);
  attn<<<agrid, 256, 0, stream>>>(Qb, Kb, Vt, out);

  row0_fix<<<BATCH, DOUT, 0, stream>>>(Vt, out);
}
